// GATv2Encoder_43396349559251
// MI455X (gfx1250) — matrix operand order repaired and checked by reading, not compiled, not run on hardware
//
#include <hip/hip_runtime.h>
#include <stdint.h>

typedef __attribute__((ext_vector_type(16))) __bf16 v16bf;
typedef __attribute__((ext_vector_type(4)))  __bf16 v4bf;
typedef __attribute__((ext_vector_type(8)))  float  v8f;

#define ED_ 32
#define HC_ 128
#define H_  4

__device__ __forceinline__ void atomicAddF(float* p, float v) {
    unsafeAtomicAdd(p, v);   // global_atomic_add_f32 on CDNA
}

// ---------------------------------------------------------------------------
// Self-loop attr = mean of incoming edge_attr  (PyG fill_value='mean')
// ---------------------------------------------------------------------------
__global__ void k_deg(const long long* __restrict__ ei, int* __restrict__ deg, int nE) {
    int e = blockIdx.x * blockDim.x + threadIdx.x;
    if (e < nE) atomicAdd(&deg[(int)ei[nE + e]], 1);
}

__global__ void k_loopsum(const long long* __restrict__ ei, const float* __restrict__ eattr,
                          float* __restrict__ la, int nE) {
    long long i = (long long)blockIdx.x * blockDim.x + threadIdx.x;
    if (i < (long long)nE * ED_) {
        long long e = i >> 5; int j = (int)(i & 31);
        int d = (int)ei[nE + e];
        atomicAddF(&la[(long long)d * ED_ + j], eattr[i]);
    }
}

__global__ void k_loopdiv(float* __restrict__ la, const int* __restrict__ deg, int nN) {
    int i = blockIdx.x * blockDim.x + threadIdx.x;
    if (i < nN * ED_) la[i] /= fmaxf((float)deg[i >> 5], 1.f);
}

// ---------------------------------------------------------------------------
// WMMA GEMM:  C[M x NT*16] = A[M x K] @ W[K x NT*16] + bias
// fp32 in/out, bf16 WMMA compute, fp32 accumulate.
// 256 threads = 8 waves; wave w owns rows [w*16, w*16+16) of a 128-row block.
// Both LDS tiles are staged in WMMA *fragment order*: lane's 16 bf16 are
// contiguous (32B), so a fragment = one v16bf LDS read (2x ds_load_b128).
//
// A-frag (16x32):  lane = (row&15) + 16*((k>>3)&1),  t = (k&7) + (k&16 ? 8:0)
// B-frag (32x16):  lane = (col&15) + 16*((k>>3)&1),  t = (k&7) + (k&16 ? 8:0)
// ---------------------------------------------------------------------------
template<int NT>
__global__ __launch_bounds__(256)
void k_gemm_bias(const float* __restrict__ A, const float* __restrict__ W,
                 const float* __restrict__ bias, float* __restrict__ Cout,
                 int M, int K) {
    __shared__ __bf16 Bs[4][NT][32][16];   // whole W, fragment order (K<=128)
    __shared__ __bf16 As[8][32][16];       // one 32-wide k-step of A, fragment order

    const int tid  = threadIdx.x;
    const int lane = tid & 31;
    const int wave = tid >> 5;
    const int row0 = blockIdx.x * 128;
    const int ncol = lane & 15;
    const int nK   = K >> 5;

    // stage entire weight tile once, fragment order
    for (int idx = tid; idx < K * NT * 16; idx += 256) {
        int r  = idx / (NT * 16);            // k index
        int c  = idx % (NT * 16);            // column
        int rr = r & 31;
        Bs[r >> 5][c >> 4][(c & 15) + (((rr >> 3) & 1) << 4)][(rr & 7) + ((rr & 16) ? 8 : 0)] = (__bf16)W[idx];
    }

    v8f acc[NT];
    #pragma unroll
    for (int nt = 0; nt < NT; ++nt) {
        float bv = bias ? bias[nt * 16 + ncol] : 0.f;
        #pragma unroll
        for (int i = 0; i < 8; ++i) acc[nt][i] = bv;
    }

    for (int ks = 0; ks < nK; ++ks) {
        // stage A k-step: 128 rows x 8 k-quads; float4 global read -> 8B DS store
        for (int idx = tid; idx < 128 * 8; idx += 256) {
            int r  = idx >> 3;
            int qd = idx & 7;
            int kk = qd << 2;
            int gr = row0 + r;
            float4 v = make_float4(0.f, 0.f, 0.f, 0.f);
            if (gr < M)
                v = *(const float4*)(A + (long long)gr * K + (ks << 5) + kk);
            v4bf pk;
            pk[0] = (__bf16)v.x; pk[1] = (__bf16)v.y;
            pk[2] = (__bf16)v.z; pk[3] = (__bf16)v.w;
            int bl = (r & 15) + (((kk >> 3) & 1) << 4);
            int t  = (kk & 7) + ((kk & 16) ? 8 : 0);
            *(v4bf*)&As[r >> 4][bl][t] = pk;
        }
        __syncthreads();

        v16bf afrag = *(const v16bf*)&As[wave][lane][0];
        #pragma unroll
        for (int nt = 0; nt < NT; ++nt) {
            v16bf bfrag = *(const v16bf*)&Bs[ks][nt][lane][0];
            acc[nt] = __builtin_amdgcn_wmma_f32_16x16x32_bf16(
                false, afrag, false, bfrag, (short)0, acc[nt], false, false);
        }
        __syncthreads();
    }

    // C/D layout: VGPR i -> row (lane<16 ? i : 8+i), col = lane&15
    int mbase = row0 + wave * 16 + ((lane < 16) ? 0 : 8);
    #pragma unroll
    for (int nt = 0; nt < NT; ++nt) {
        int c = nt * 16 + ncol;
        #pragma unroll
        for (int i = 0; i < 8; ++i) {
            int m = mbase + i;
            if (m < M) Cout[(long long)m * (NT * 16) + c] = acc[nt][i];
        }
    }
}

// ---------------------------------------------------------------------------
// Per-edge attention logits + segment max.  One wave32 per edge.
// e = leaky_relu(xl[src] + xr[dst] + ea@We); alpha[h] = <e_h, att_h>
// ---------------------------------------------------------------------------
__global__ __launch_bounds__(256)
void k_alpha(const float* __restrict__ xl, const float* __restrict__ xr,
             const float* __restrict__ We, const float* __restrict__ att,
             const long long* __restrict__ ei, const float* __restrict__ eattr,
             const float* __restrict__ la,
             float* __restrict__ alpha, unsigned* __restrict__ amax_u,
             int nE, int nN) {
    __shared__ float WeS[ED_ * HC_];
    __shared__ float attS[HC_];
    for (int i = threadIdx.x; i < ED_ * HC_; i += blockDim.x) WeS[i] = We[i];
    for (int i = threadIdx.x; i < HC_;       i += blockDim.x) attS[i] = att[i];
    __syncthreads();

    const int lane = threadIdx.x & 31;
    const int wave = threadIdx.x >> 5;
    const long long Ea = (long long)nE + nN;
    const long long wstride = (long long)gridDim.x * 8;

    for (long long e = (long long)blockIdx.x * 8 + wave; e < Ea; e += wstride) {
        int s, d; const float* ea;
        if (e < nE) { s = (int)ei[e]; d = (int)ei[nE + e]; ea = eattr + e * ED_; }
        else        { s = d = (int)(e - nE);               ea = la + (long long)s * ED_; }

        float eav = ea[lane];
        float acc[H_] = {0.f, 0.f, 0.f, 0.f};
        for (int k = 0; k < ED_; ++k) {
            float a = __shfl(eav, k, 32);
            const float* wr = &WeS[k * HC_];
            #pragma unroll
            for (int t = 0; t < H_; ++t) acc[t] += a * wr[lane + 32 * t];
        }
        float p[H_];
        #pragma unroll
        for (int t = 0; t < H_; ++t) {
            int j = lane + 32 * t;                      // head = t (channel block)
            float v = xl[(long long)s * HC_ + j] + xr[(long long)d * HC_ + j] + acc[t];
            v = (v > 0.f) ? v : 0.2f * v;               // leaky_relu
            p[t] = v * attS[j];
        }
        #pragma unroll
        for (int t = 0; t < H_; ++t)
            #pragma unroll
            for (int off = 16; off > 0; off >>= 1)
                p[t] += __shfl_xor(p[t], off, 32);
        if (lane == 0) {
            #pragma unroll
            for (int t = 0; t < H_; ++t) {
                alpha[e * H_ + t] = p[t];
                int b = __float_as_int(p[t]);
                unsigned key = (b >= 0) ? ((unsigned)b | 0x80000000u) : (unsigned)~b;
                atomicMax(&amax_u[d * H_ + t], key);    // monotone float ordering
            }
        }
    }
}

// exp(alpha - amax) and segment denominator
__global__ void k_exp(const long long* __restrict__ ei, float* __restrict__ alpha,
                      const unsigned* __restrict__ amax_u, float* __restrict__ denom,
                      int nE, int nN) {
    long long tot = ((long long)nE + nN) * H_;
    for (long long i = (long long)blockIdx.x * blockDim.x + threadIdx.x; i < tot;
         i += (long long)gridDim.x * blockDim.x) {
        long long e = i >> 2; int h = (int)(i & 3);
        int d = (e < nE) ? (int)ei[nE + e] : (int)(e - nE);
        unsigned k = amax_u[d * H_ + h];
        int b = (k & 0x80000000u) ? (int)(k ^ 0x80000000u) : (int)~k;
        float am = __int_as_float(b);
        if (!(fabsf(am) < 3.0e38f)) am = 0.f;           // !isfinite -> 0 (matches ref)
        float ex = __expf(alpha[i] - am);
        alpha[i] = ex;
        atomicAddF(&denom[d * H_ + h], ex);
    }
}

// scatter-add:  agg[dst] += xl[src] * softmax_weight
__global__ __launch_bounds__(256)
void k_agg(const float* __restrict__ xl, const float* __restrict__ alpha,
           const float* __restrict__ denom, const long long* __restrict__ ei,
           float* __restrict__ agg, int nE, int nN) {
    const int lane = threadIdx.x & 31;
    const int wave = threadIdx.x >> 5;
    const long long Ea = (long long)nE + nN;
    for (long long e = (long long)blockIdx.x * 8 + wave; e < Ea;
         e += (long long)gridDim.x * 8) {
        int s, d;
        if (e < nE) { s = (int)ei[e]; d = (int)ei[nE + e]; }
        else        { s = d = (int)(e - nE); }
        #pragma unroll
        for (int t = 0; t < H_; ++t) {
            float w = alpha[e * H_ + t] / (denom[d * H_ + t] + 1e-16f);
            int j = lane + 32 * t;
            atomicAddF(&agg[(long long)d * HC_ + j], xl[(long long)s * HC_ + j] * w);
        }
    }
}

__global__ void k_bias_relu(const float* __restrict__ agg, const float* __restrict__ bias,
                            float* __restrict__ out, int nN) {
    int i = blockIdx.x * blockDim.x + threadIdx.x;
    if (i < nN * HC_) out[i] = fmaxf(agg[i] + bias[i & (HC_ - 1)], 0.f);
}

// ---------------------------------------------------------------------------
extern "C" void kernel_launch(void* const* d_in, const int* in_sizes, int n_in,
                              void* d_out, int out_size, void* d_ws, size_t ws_size,
                              hipStream_t stream) {
    const float*     x     = (const float*)d_in[0];
    const long long* ei    = (const long long*)d_in[1];   // int64 per reference
    const float*     eattr = (const float*)d_in[2];
    const float *W1l = (const float*)d_in[3],  *b1l = (const float*)d_in[4];
    const float *W1r = (const float*)d_in[5],  *b1r = (const float*)d_in[6];
    const float *W1e = (const float*)d_in[7],  *att1 = (const float*)d_in[8];
    const float *bias1 = (const float*)d_in[9];
    const float *W2l = (const float*)d_in[10], *b2l = (const float*)d_in[11];
    const float *W2r = (const float*)d_in[12], *b2r = (const float*)d_in[13];
    const float *W2e = (const float*)d_in[14], *att2 = (const float*)d_in[15];
    const float *bias2 = (const float*)d_in[16];
    const float *Wo1 = (const float*)d_in[17], *bo1 = (const float*)d_in[18];
    const float *Wo2 = (const float*)d_in[19], *bo2 = (const float*)d_in[20];
    float* out = (float*)d_out;

    const int N = in_sizes[0] / HC_;     // FIN == HC == 128
    const int E = in_sizes[1] / 2;
    const long long Ea = (long long)E + N;

    // workspace carve-out (~100 MB)
    uintptr_t p = (uintptr_t)d_ws;
    auto alloc = [&](size_t bytes) { uintptr_t q = p; p += (bytes + 255) & ~(size_t)255; return (void*)q; };
    int*      deg   = (int*)     alloc((size_t)N * 4);
    float*    la    = (float*)   alloc((size_t)N * ED_ * 4);   // loop_attr, reused as MLP temp
    float*    xl    = (float*)   alloc((size_t)N * HC_ * 4);
    float*    xr    = (float*)   alloc((size_t)N * HC_ * 4);
    float*    alpha = (float*)   alloc((size_t)Ea * H_ * 4);
    unsigned* amax  = (unsigned*)alloc((size_t)N * H_ * 4);
    float*    denom = (float*)   alloc((size_t)N * H_ * 4);
    float*    agg   = (float*)   alloc((size_t)N * HC_ * 4);
    float*    h     = (float*)   alloc((size_t)N * HC_ * 4);

    // self-loop attributes
    hipMemsetAsync(deg, 0, (size_t)N * 4, stream);
    hipMemsetAsync(la,  0, (size_t)N * ED_ * 4, stream);
    k_deg    <<<(E + 255) / 256, 256, 0, stream>>>(ei, deg, E);
    k_loopsum<<<(unsigned)(((long long)E * ED_ + 255) / 256), 256, 0, stream>>>(ei, eattr, la, E);
    k_loopdiv<<<(N * ED_ + 255) / 256, 256, 0, stream>>>(la, deg, N);

    const dim3 gg((N + 127) / 128);
    auto conv = [&](const float* xin, const float* Wl, const float* bl,
                    const float* Wr, const float* br, const float* We,
                    const float* att, const float* bias, float* hout) {
        hipMemsetAsync(amax,  0, (size_t)N * H_ * 4, stream);
        hipMemsetAsync(denom, 0, (size_t)N * H_ * 4, stream);
        hipMemsetAsync(agg,   0, (size_t)N * HC_ * 4, stream);
        k_gemm_bias<8><<<gg, 256, 0, stream>>>(xin, Wl, bl, xl, N, HC_);
        k_gemm_bias<8><<<gg, 256, 0, stream>>>(xin, Wr, br, xr, N, HC_);
        k_alpha<<<2048, 256, 0, stream>>>(xl, xr, We, att, ei, eattr, la, alpha, amax, E, N);
        k_exp<<<(unsigned)((Ea * H_ + 255) / 256), 256, 0, stream>>>(ei, alpha, amax, denom, E, N);
        k_agg<<<(unsigned)((Ea + 7) / 8), 256, 0, stream>>>(xl, alpha, denom, ei, agg, E, N);
        k_bias_relu<<<(N * HC_ + 255) / 256, 256, 0, stream>>>(agg, bias, hout, N);
    };

    conv(x, W1l, b1l, W1r, b1r, W1e, att1, bias1, h);
    conv(h, W2l, b2l, W2r, b2r, W2e, att2, bias2, h);   // in-place: h consumed by GEMMs first

    // output MLP: (h @ Wo1 + bo1) @ Wo2 + bo2 ; reuse `la` for the N x 32 temp
    float* t1 = la;
    k_gemm_bias<2><<<gg, 256, 0, stream>>>(h,  Wo1, bo1, t1,  N, HC_);
    k_gemm_bias<4><<<gg, 256, 0, stream>>>(t1, Wo2, bo2, out, N, ED_);
}
